// self_attention_61392262529323
// MI455X (gfx1250) — matrix operand order repaired and checked by reading, not compiled, not run on hardware
//
#include <hip/hip_runtime.h>

// Problem dims (fixed by reference): T=256 timesteps, N=512 nodes, D=H=O=512.
#define T_DIM 256
#define N_DIM 512
#define D_DIM 512
#define H_DIM 512

typedef __attribute__((ext_vector_type(16))) __bf16          v16bf;
typedef __attribute__((ext_vector_type(8)))  float           v8f;
typedef __attribute__((ext_vector_type(4)))  float           f32x4;
typedef __attribute__((ext_vector_type(4)))  unsigned int    u32x4;
typedef __attribute__((ext_vector_type(4)))  unsigned short  u16x4;

typedef __attribute__((address_space(3))) unsigned short lds_us;

union Frag {
  v16bf bf;
  u32x4 q[2];
};
static_assert(sizeof(Frag) == 32, "frag is 8 VGPRs");

__device__ __forceinline__ unsigned short f2bf(float f) {
  unsigned int u = __float_as_uint(f);
  return (unsigned short)((u + 0x7fffu + ((u >> 16) & 1u)) >> 16); // RNE
}

__device__ __forceinline__ v8f wmma_bf16(const Frag& a, const Frag& b, v8f c) {
  return __builtin_amdgcn_wmma_f32_16x16x32_bf16(false, a.bf, false, b.bf,
                                                 (short)0, c, false, false);
}

// CDNA5 async global->LDS copy, 16B per lane (GVS mode: saddr + 32-bit voffset).
__device__ __forceinline__ void async_g2l_b128(unsigned lds_byte_addr,
                                               unsigned gbyte_off,
                                               unsigned long long sbase) {
  asm volatile("global_load_async_to_lds_b128 %0, %1, %2"
               :: "v"(lds_byte_addr), "v"(gbyte_off), "s"(sbase)
               : "memory");
}
__device__ __forceinline__ void wait_async0() {
  asm volatile("s_wait_asynccnt 0" ::: "memory");
}
__device__ __forceinline__ unsigned lds_addr_of(const unsigned short* p) {
  return (unsigned)(size_t)(lds_us*)p;
}

// ---------------------------------------------------------------------------
// Pre-pass A: X f32 -> bf16 (one shot; removes 12x redundant conversions).
// ---------------------------------------------------------------------------
__global__ __launch_bounds__(256) void cvt_x_kernel(
    const float* __restrict__ X, unsigned short* __restrict__ Xb) {
  size_t i = ((size_t)blockIdx.x * 256 + threadIdx.x) * 8;
  f32x4 a = *(const f32x4*)(X + i);
  f32x4 b = *(const f32x4*)(X + i + 4);
  u16x4 o0, o1;
  o0.x = f2bf(a.x); o0.y = f2bf(a.y); o0.z = f2bf(a.z); o0.w = f2bf(a.w);
  o1.x = f2bf(b.x); o1.y = f2bf(b.y); o1.z = f2bf(b.z); o1.w = f2bf(b.w);
  *(u16x4*)(Xb + i)     = o0;
  *(u16x4*)(Xb + i + 4) = o1;
}

// ---------------------------------------------------------------------------
// Pre-pass B: W[512x512] f32 -> Wt[h][d] bf16 (transpose via LDS 64x64 tile).
// ---------------------------------------------------------------------------
__global__ __launch_bounds__(256) void cvt_wt_kernel(
    const float* __restrict__ W, unsigned short* __restrict__ Wt) {
  __shared__ float tile[64][65];
  const int tx = threadIdx.x & 63;
  const int ty = threadIdx.x >> 6;
  const int d0 = blockIdx.x * 64;
  const int h0 = blockIdx.y * 64;
  for (int j = 0; j < 16; ++j) {
    int row = ty + j * 4;
    tile[row][tx] = W[(size_t)(d0 + row) * H_DIM + h0 + tx];
  }
  __syncthreads();
  for (int j = 0; j < 16; ++j) {
    int row = ty + j * 4;
    Wt[(size_t)(h0 + row) * D_DIM + d0 + tx] = f2bf(tile[tx][row]);
  }
}

// ---------------------------------------------------------------------------
// Kernel 1: Xb[131072 x 512] bf16 @ Wt (stored [h][d]) -> bf16.
// Double-buffered async staging: issue buf^1 -> WMMA buf -> wait -> barrier.
// vmode==0: store [node][t][h] (Q,K);  vmode==1: store [node][h][t] (V).
// ---------------------------------------------------------------------------
__global__ __launch_bounds__(256) void qkv_gemm_kernel(
    const unsigned short* __restrict__ Xb, const unsigned short* __restrict__ Wt,
    unsigned short* __restrict__ outB, int vmode) {
  __shared__ unsigned short As[2][128 * 64];  // 2 x 16 KB
  __shared__ unsigned short Bs[2][128 * 64];  // 2 x 16 KB

  const int tid  = threadIdx.x;
  const int lane = tid & 31;
  const int wave = tid >> 5;
  const int wm   = wave & 3;   // 32-row strip
  const int wn   = wave >> 2;  // 64-col half
  const int lr   = lane & 15;
  const int hi   = lane >> 4;
  const int m0   = blockIdx.x * 128;
  const int n0   = blockIdx.y * 128;

  const unsigned long long xbase = (unsigned long long)Xb;
  const unsigned long long wbase = (unsigned long long)Wt;
  const unsigned asA = lds_addr_of(&As[0][0]);
  const unsigned asB = lds_addr_of(&Bs[0][0]);

  auto stage = [&](int buf, int kt) {  // 128x64 bf16 per matrix, 4 b128/thread
    for (int j = 0; j < 4; ++j) {
      int idx = tid + j * 256;
      int row = idx >> 3;          // 0..127
      int kc  = (idx & 7) << 3;    // 0,8,..,56
      unsigned loff = (unsigned)buf * 16384u + (unsigned)(row * 64 + kc) * 2;
      async_g2l_b128(asA + loff,
                     (unsigned)(((m0 + row) * D_DIM + kt * 64 + kc) * 2), xbase);
      async_g2l_b128(asB + loff,
                     (unsigned)(((n0 + row) * D_DIM + kt * 64 + kc) * 2), wbase);
    }
  };

  v8f zero = {};
  v8f acc[2][4];
  for (int i = 0; i < 2; ++i)
    for (int j = 0; j < 4; ++j) acc[i][j] = zero;

  stage(0, 0);
  wait_async0();
  __syncthreads();

  for (int kt = 0; kt < D_DIM / 64; ++kt) {  // 8 iterations
    int cur = kt & 1;
    if (kt + 1 < D_DIM / 64) stage(cur ^ 1, kt + 1);  // overlap with compute

    const unsigned short* Ab = &As[cur][0];
    const unsigned short* Bb = &Bs[cur][0];
    for (int ks = 0; ks < 2; ++ks) {
      Frag a[2];
      for (int rt = 0; rt < 2; ++rt) {
        int row  = wm * 32 + rt * 16 + lr;
        int base = row * 64 + ks * 32 + (hi ? 8 : 0);
        a[rt].q[0] = *(const u32x4*)&Ab[base];
        a[rt].q[1] = *(const u32x4*)&Ab[base + 16];
      }
      for (int ct = 0; ct < 4; ++ct) {
        Frag b;
        int col = wn * 64 + ct * 16 + lr;
        int kb  = col * 64 + ks * 32 + (hi ? 8 : 0);
        b.q[0] = *(const u32x4*)&Bb[kb];
        b.q[1] = *(const u32x4*)&Bb[kb + 16];
        for (int rt = 0; rt < 2; ++rt) acc[rt][ct] = wmma_bf16(a[rt], b, acc[rt][ct]);
      }
    }
    wait_async0();   // next tile's DMA has had the whole compute phase to land
    __syncthreads();
  }

  if (vmode == 0) {
    for (int rt = 0; rt < 2; ++rt)
      for (int ct = 0; ct < 4; ++ct)
        for (int r = 0; r < 8; ++r) {
          int m    = m0 + wm * 32 + rt * 16 + r + hi * 8;  // m = t*N + node
          int h    = n0 + wn * 64 + ct * 16 + lr;
          int t    = m >> 9;
          int node = m & (N_DIM - 1);
          outB[((size_t)node * T_DIM + t) * H_DIM + h] = f2bf(acc[rt][ct][r]);
        }
  } else {
    for (int rt = 0; rt < 2; ++rt)
      for (int ct = 0; ct < 4; ++ct)
        for (int r = 0; r < 8; ++r) {
          int m    = m0 + wm * 32 + rt * 16 + r + hi * 8;
          int h    = n0 + wn * 64 + ct * 16 + lr;
          int t    = m >> 9;
          int node = m & (N_DIM - 1);
          outB[((size_t)node * H_DIM + h) * T_DIM + t] = f2bf(acc[rt][ct][r]);
        }
  }
}

// ---------------------------------------------------------------------------
// Kernel 2: per-node attention, one workgroup = (node, 64-query block).
// Phase 1: Q/K tiles async-staged (double buffer) -> S in regs -> softmax.
// Phase 2: V^T tiles async-staged (double buffer) -> out = P V.
// All tiles staged once per WG (removes 4-5x redundant global reads).
// ---------------------------------------------------------------------------
__global__ __launch_bounds__(256) void attn_kernel(
    const unsigned short* __restrict__ Qb,   // [N][T][H] bf16
    const unsigned short* __restrict__ Kb,   // [N][T][H] bf16
    const unsigned short* __restrict__ Vt,   // [N][O][T] bf16 (transposed)
    const int* __restrict__ mask,            // [T]
    float* __restrict__ out) {               // [T][N][O] f32
  __shared__ unsigned short Pl[64 * 256];      // softmaxed P, 32 KB
  __shared__ unsigned short Stage[2][512 * 32];// 2 x 32 KB staging
  __shared__ float redmax[2][64];
  __shared__ float redsum[2][64];

  const int tid  = threadIdx.x;
  const int lane = tid & 31;
  const int wave = tid >> 5;
  const int lr   = lane & 15;
  const int hi   = lane >> 4;
  const int qb   = blockIdx.x;   // 64-row query block
  const int node = blockIdx.y;

  const unsigned long long qbase =
      (unsigned long long)(Qb + (size_t)node * T_DIM * H_DIM);
  const unsigned long long kbase =
      (unsigned long long)(Kb + (size_t)node * T_DIM * H_DIM);
  const unsigned long long vbase =
      (unsigned long long)(Vt + (size_t)node * H_DIM * T_DIM);
  const unsigned asS = lds_addr_of(&Stage[0][0]);
  const unsigned bufStride = 512u * 32u * 2u;  // bytes per staging buffer

  const float scale = 0.04419417382415922f;  // 1/sqrt(512)

  const int rs = wave >> 1;  // 16-row strip of the 64-row query block
  const int ch = wave & 1;   // 128-key half

  // Phase-1 stager: K tile 256x32 (u16 idx 0..8191) + Q tile 64x32 (8192..).
  auto stage1 = [&](int buf, int kk) {
    unsigned b = asS + (unsigned)buf * bufStride;
    for (int j = 0; j < 4; ++j) {
      int idx = tid + j * 256;
      int row = idx >> 2;          // key row 0..255
      int kc  = (idx & 3) << 3;
      async_g2l_b128(b + (unsigned)(row * 32 + kc) * 2,
                     (unsigned)((row * H_DIM + kk + kc) * 2), kbase);
    }
    int row = tid >> 2;            // query row 0..63
    int kc  = (tid & 3) << 3;
    async_g2l_b128(b + (unsigned)(8192 + row * 32 + kc) * 2,
                   (unsigned)(((qb * 64 + row) * H_DIM + kk + kc) * 2), qbase);
  };

  v8f zero = {};
  v8f sacc[8];
  for (int i = 0; i < 8; ++i) sacc[i] = zero;

  stage1(0, 0);
  wait_async0();
  __syncthreads();

  for (int kt = 0; kt < H_DIM / 32; ++kt) {  // 16 iterations
    int cur = kt & 1;
    if (kt + 1 < H_DIM / 32) stage1(cur ^ 1, (kt + 1) * 32);

    const unsigned short* Sb = &Stage[cur][0];
    Frag a;
    {
      int qrow = rs * 16 + lr;
      int ka   = hi ? 8 : 0;
      a.q[0] = *(const u32x4*)&Sb[8192 + qrow * 32 + ka];
      a.q[1] = *(const u32x4*)&Sb[8192 + qrow * 32 + ka + 16];
    }
    for (int ct = 0; ct < 8; ++ct) {
      Frag b;
      int scol = ch * 128 + ct * 16 + lr;
      int kb2  = hi ? 8 : 0;
      b.q[0] = *(const u32x4*)&Sb[scol * 32 + kb2];
      b.q[1] = *(const u32x4*)&Sb[scol * 32 + kb2 + 16];
      sacc[ct] = wmma_bf16(a, b, sacc[ct]);
    }
    wait_async0();
    __syncthreads();
  }

  // ---------------- softmax over the full 256-key row ----------------
  int mk[8];
  for (int ct = 0; ct < 8; ++ct) mk[ct] = mask[ch * 128 + ct * 16 + lr];

  float pm[8];
  for (int r = 0; r < 8; ++r) pm[r] = -1e30f;
  for (int ct = 0; ct < 8; ++ct)
    for (int r = 0; r < 8; ++r) {
      float v = mk[ct] ? sacc[ct][r] * scale : -1e9f;
      sacc[ct][r] = v;
      pm[r] = fmaxf(pm[r], v);
    }
  for (int r = 0; r < 8; ++r)      // xor 1/2/4/8 stays inside 16-lane halves
    for (int off = 1; off <= 8; off <<= 1)
      pm[r] = fmaxf(pm[r], __shfl_xor(pm[r], off, 32));

  const int mrow0 = rs * 16 + hi * 8;
  if (lr == 0)
    for (int r = 0; r < 8; ++r) redmax[ch][mrow0 + r] = pm[r];
  __syncthreads();

  float rm[8];
  for (int r = 0; r < 8; ++r)
    rm[r] = fmaxf(redmax[0][mrow0 + r], redmax[1][mrow0 + r]);

  float ps[8];
  for (int r = 0; r < 8; ++r) ps[r] = 0.f;
  for (int ct = 0; ct < 8; ++ct)
    for (int r = 0; r < 8; ++r) {
      float e = __expf(sacc[ct][r] - rm[r]);
      sacc[ct][r] = e;
      ps[r] += e;
    }
  for (int r = 0; r < 8; ++r)
    for (int off = 1; off <= 8; off <<= 1)
      ps[r] += __shfl_xor(ps[r], off, 32);
  if (lr == 0)
    for (int r = 0; r < 8; ++r) redsum[ch][mrow0 + r] = ps[r];
  __syncthreads();

  for (int r = 0; r < 8; ++r) {
    float rinv = 1.0f / (redsum[0][mrow0 + r] + redsum[1][mrow0 + r]);
    for (int ct = 0; ct < 8; ++ct) {
      int s = ch * 128 + ct * 16 + lr;
      Pl[(mrow0 + r) * 256 + s] = f2bf(sacc[ct][r] * rinv);
    }
  }
  __syncthreads();

  // ---------------- Phase 2: out = P V ----------------
  const int rs2 = wave >> 1;
  const int oh  = wave & 1;

  auto stage2 = [&](int buf, int sc) {  // V^T tile 512x32 = 32 KB
    unsigned b = asS + (unsigned)buf * bufStride;
    for (int j = 0; j < 8; ++j) {
      int idx = tid + j * 256;
      int row = idx >> 2;          // o row 0..511
      int kc  = (idx & 3) << 3;
      async_g2l_b128(b + (unsigned)(row * 32 + kc) * 2,
                     (unsigned)((row * T_DIM + sc * 32 + kc) * 2), vbase);
    }
  };

  v8f oacc[16];
  for (int i = 0; i < 16; ++i) oacc[i] = zero;

  stage2(0, 0);
  wait_async0();
  __syncthreads();

  for (int sc = 0; sc < 8; ++sc) {
    int cur = sc & 1;
    if (sc + 1 < 8) stage2(cur ^ 1, sc + 1);

    const unsigned short* Sb = &Stage[cur][0];
    Frag a;
    {
      int prow = rs2 * 16 + lr;
      int sa   = sc * 32 + (hi ? 8 : 0);
      a.q[0] = *(const u32x4*)&Pl[prow * 256 + sa];
      a.q[1] = *(const u32x4*)&Pl[prow * 256 + sa + 16];
    }
    for (int ot = 0; ot < 16; ++ot) {
      Frag b;
      int o  = oh * 256 + ot * 16 + lr;
      int sb = hi ? 8 : 0;
      b.q[0] = *(const u32x4*)&Sb[o * 32 + sb];
      b.q[1] = *(const u32x4*)&Sb[o * 32 + sb + 16];
      oacc[ot] = wmma_bf16(a, b, oacc[ot]);
    }
    wait_async0();
    __syncthreads();
  }

  for (int ot = 0; ot < 16; ++ot)
    for (int r = 0; r < 8; ++r) {
      int t = qb * 64 + rs2 * 16 + r + hi * 8;
      int o = oh * 256 + ot * 16 + lr;
      out[((size_t)t * N_DIM + node) * H_DIM + o] = oacc[ot][r];  // coalesced
    }
}

// ---------------------------------------------------------------------------
extern "C" void kernel_launch(void* const* d_in, const int* in_sizes, int n_in,
                              void* d_out, int out_size, void* d_ws,
                              size_t ws_size, hipStream_t stream) {
  (void)in_sizes; (void)n_in; (void)out_size; (void)ws_size;
  const float* X    = (const float*)d_in[0];  // [T,N,D]
  const float* Wq   = (const float*)d_in[1];
  const float* Wk   = (const float*)d_in[2];
  const float* Wv   = (const float*)d_in[3];
  const int*   mask = (const int*)d_in[4];    // [1,1,T]
  float*       out  = (float*)d_out;          // [T,N,O]

  const size_t xelems = (size_t)T_DIM * N_DIM * D_DIM;  // 67,108,864
  const size_t welems = (size_t)D_DIM * H_DIM;          // 262,144
  unsigned short* Xb  = (unsigned short*)d_ws;
  unsigned short* Wtq = Xb + xelems;
  unsigned short* Wtk = Wtq + welems;
  unsigned short* Wtv = Wtk + welems;
  unsigned short* Qw  = Wtv + welems;
  unsigned short* Kw  = Qw + xelems;
  unsigned short* Vw  = Kw + xelems;

  cvt_x_kernel<<<dim3(xelems / (256 * 8)), 256, 0, stream>>>(X, Xb);
  dim3 gt(D_DIM / 64, H_DIM / 64);
  cvt_wt_kernel<<<gt, 256, 0, stream>>>(Wq, Wtq);
  cvt_wt_kernel<<<gt, 256, 0, stream>>>(Wk, Wtk);
  cvt_wt_kernel<<<gt, 256, 0, stream>>>(Wv, Wtv);

  dim3 g1((T_DIM * N_DIM) / 128, H_DIM / 128);  // 1024 x 4
  qkv_gemm_kernel<<<g1, 256, 0, stream>>>(Xb, Wtq, Qw, 0);
  qkv_gemm_kernel<<<g1, 256, 0, stream>>>(Xb, Wtk, Kw, 0);
  qkv_gemm_kernel<<<g1, 256, 0, stream>>>(Xb, Wtv, Vw, 1);

  dim3 g2(T_DIM / 64, N_DIM);  // 4 x 512
  attn_kernel<<<g2, 256, 0, stream>>>(Qw, Kw, Vw, mask, out);
}
